// AminoSubBlock_57707180589709
// MI455X (gfx1250) — matrix operand order repaired and checked by reading, not compiled, not run on hardware
//
#include <hip/hip_runtime.h>
#include <hip/hip_bf16.h>
#include <stdint.h>

// ---------------------------------------------------------------------------
// Problem constants (match reference)
// ---------------------------------------------------------------------------
#define B_DIM   64
#define NRES    256
#define A_ATOMS 8
#define DIM_A   128
#define DIM_V   64
#define CHAN    128
#define MROWS   (B_DIM * NRES)      // 16384 residue rows
#define NATOMS  (NRES * A_ATOMS)    // 2048

typedef __attribute__((ext_vector_type(16))) _Float16 v16h;
typedef __attribute__((ext_vector_type(8)))  _Float16 v8h;
typedef __attribute__((ext_vector_type(8)))  float    v8f;

// ---------------------------------------------------------------------------
// WMMA fragment helpers (wave32; C per CDNA5 ISA §7.12.2, A/B use contiguous k, not the ISA table order)
//   A 16x32 f16 : lane m=lane&15, half=lane>>4; element e:
//      k = half*16 + e (contiguous, matches B)
//   B 32x16 f16 : lane n=lane&15; element e -> k = (lane>>4)*16 + e (contiguous)
//   C 16x16 f32 : VGPR r, lane -> (m = r + 8*(lane>>4), n = lane&15)
// ---------------------------------------------------------------------------
__device__ __forceinline__ v16h load_a_frag(const _Float16* tile, int ldA, int k0, int lane) {
    int m = lane & 15, half = lane >> 4;
    const _Float16* p = tile + m * ldA + k0 + half * 16;
    v8h lo = *(const v8h*)(p);        // k0+half*16 .. +7
    v8h hi = *(const v8h*)(p + 8);    // k0+half*16+8 .. +15
    v16h a;
#pragma unroll
    for (int e = 0; e < 8; ++e) { a[e] = lo[e]; a[e + 8] = hi[e]; }
    return a;
}

__device__ __forceinline__ v16h load_b_frag(const _Float16* W, int ldW, int n0, int k0, int lane) {
    int n = lane & 15;
    int koff = (lane >> 4) * 16;
    return *(const v16h*)(W + (long long)(n0 + n) * ldW + k0 + koff);
}

__device__ __forceinline__ v8f wmma32(v16h a, v16h b, v8f c) {
    return __builtin_amdgcn_wmma_f32_16x16x32_f16(false, a, false, b, (short)0, c, false, false);
}

__device__ __forceinline__ void store_tile_h(_Float16* buf, int ldB, v8f acc, int n0, int lane) {
    int n = lane & 15, hs = lane >> 4;
#pragma unroll
    for (int r = 0; r < 8; ++r) buf[(r + 8 * hs) * ldB + n0 + n] = (_Float16)acc[r];
}

// ---------------------------------------------------------------------------
// Generic WMMA GEMM:  C = alpha * (A f32 -> f16) @ W^T (+bias, leaky, +resid)
// A element (m,k) at A[aBase + y*aY + z*aZ + m*aRow + k*aCol]   (f32, strided)
// W row-major [N][K] f16, tile offset y*wY
// One block = 16 rows, 8 waves; wave w handles N-tile(s) w, w+8, ...
// ---------------------------------------------------------------------------
__launch_bounds__(256)
__global__ void gemm_kernel(
    const float* __restrict__ A, long long aBase, int aRow, int aCol, long long aY, long long aZ,
    const _Float16* __restrict__ W, long long wY,
    float* __restrict__ C, long long cBase, int cRow, int cCol, long long cY, long long cZ,
    const float* __restrict__ bias,
    const float* __restrict__ R, long long rBase, int rRow, int rCol, long long rY, long long rZ,
    int M, int N, int K, float alpha, int doLeaky)
{
    __shared__ __align__(32) _Float16 As[16 * 1024];
    int tid = threadIdx.x;
    int m0 = blockIdx.x * 16;
    long long y = blockIdx.y, z = blockIdx.z;
    const float* Ap = A + aBase + y * aY + z * aZ;

    for (int idx = tid; idx < 16 * K; idx += 256) {
        int m = idx / K, k = idx - m * K;
        float v = (m0 + m < M) ? Ap[(long long)(m0 + m) * aRow + (long long)k * aCol] : 0.f;
        As[m * K + k] = (_Float16)v;
    }
    __syncthreads();

    int wave = tid >> 5, lane = tid & 31;
    const _Float16* Wp = W + y * wY;
    for (int nt = wave; nt * 16 < N; nt += 8) {
        int n0 = nt * 16;
        v8f acc = {};
        for (int k0 = 0; k0 < K; k0 += 32) {
            __builtin_prefetch(Wp + (long long)n0 * K + k0 + 256, 0, 1);
            v16h a = load_a_frag(As, K, k0, lane);
            v16h b = load_b_frag(Wp, K, n0, k0, lane);
            acc = wmma32(a, b, acc);
        }
        int n = lane & 15, hs = lane >> 4;
#pragma unroll
        for (int r = 0; r < 8; ++r) {
            int m = r + 8 * hs;
            long long row = m0 + m;
            if (row >= M) continue;
            float v = alpha * acc[r];
            if (bias) v += bias[n0 + n];
            if (doLeaky) v = v > 0.f ? v : 0.1f * v;
            if (R) v += R[rBase + y * rY + z * rZ + row * (long long)rRow + (long long)(n0 + n) * rCol];
            C[cBase + y * cY + z * cZ + row * (long long)cRow + (long long)(n0 + n) * cCol] = v;
        }
    }
}

// ---------------------------------------------------------------------------
// 1-D 'SAME' conv as WMMA GEMM over an LDS halo window.
// Weight pre-packed as [N][Kw*Cin] with inner order (k, ci) so that
// A[m][k*Cin+ci] == halo[(m+k)*Cin + ci] is a contiguous sliding window.
// blockIdx = (pos_tile, batch, comp)
// ---------------------------------------------------------------------------
__launch_bounds__(256)
__global__ void conv_kernel(
    const float* __restrict__ X, int xRow, int xCol, long long xB, long long xZ,
    const _Float16* __restrict__ W,
    float* __restrict__ C, int cRow, int cCol, long long cB, long long cZ,
    const float* __restrict__ bias,
    const float* __restrict__ R, int rRow, int rCol, long long rB, long long rZ,
    int npos, int Cin, int Kw, int N, int doLeaky)
{
    __shared__ __align__(32) _Float16 As[24 * 128];   // (16+Kw-1) x Cin, Kw<=9, Cin<=128
    int tid = threadIdx.x;
    int p0 = blockIdx.x * 16;
    long long b = blockIdx.y, z = blockIdx.z;
    int pad = Kw >> 1;
    int rows = 16 + Kw - 1;
    const float* Xp = X + b * xB + z * xZ;

    for (int idx = tid; idx < rows * Cin; idx += 256) {
        int rr = idx / Cin, c = idx - rr * Cin;
        int pos = p0 - pad + rr;
        float v = (pos >= 0 && pos < npos) ? Xp[(long long)pos * xRow + (long long)c * xCol] : 0.f;
        As[rr * Cin + c] = (_Float16)v;
    }
    __syncthreads();

    int wave = tid >> 5, lane = tid & 31;
    int K = Kw * Cin;
    for (int nt = wave; nt * 16 < N; nt += 8) {
        int n0 = nt * 16;
        v8f acc = {};
        for (int k0 = 0; k0 < K; k0 += 32) {
            v16h a  = load_a_frag(As, Cin, k0, lane);   // sliding window: ldA = Cin
            v16h bb = load_b_frag(W, K, n0, k0, lane);
            acc = wmma32(a, bb, acc);
        }
        int n = lane & 15, hs = lane >> 4;
#pragma unroll
        for (int r = 0; r < 8; ++r) {
            int m = r + 8 * hs;
            int pos = p0 + m;
            if (pos >= npos) continue;
            float v = acc[r];
            if (bias) v += bias[n0 + n];
            if (doLeaky) v = v > 0.f ? v : 0.1f * v;
            if (R) v += R[rB * b + rZ * z + (long long)pos * rRow + (long long)(n0 + n) * rCol];
            C[cB * b + cZ * z + (long long)pos * cRow + (long long)(n0 + n) * cCol] = v;
        }
    }
}

// ---------------------------------------------------------------------------
// Fused self_tens_prods: per block = 16 tokens, entire chain in LDS.
//   l0/r0/l1/r1 -> y000/y110 -> y_a GEMMs -> MLP -> upd -> y011/y101/y111
//   -> y_v GEMMs -> group_norm0/group_norm1 -> residual outputs.
// ---------------------------------------------------------------------------
__launch_bounds__(256)
__global__ void tp_kernel(
    const float* __restrict__ ya2,   // [M][128]  (post conv residual)
    const float* __restrict__ yv2,   // bndi: row*192 + c*3 + i
    const _Float16* __restrict__ L0w, const _Float16* __restrict__ R0w,
    const _Float16* __restrict__ L1w, const _Float16* __restrict__ R1w,
    const _Float16* __restrict__ updLw, const _Float16* __restrict__ updRw,
    const _Float16* __restrict__ Y000w, const _Float16* __restrict__ Y110w,
    const _Float16* __restrict__ Y011w, const _Float16* __restrict__ Y101w,
    const _Float16* __restrict__ Y111w,
    const _Float16* __restrict__ w1w, const _Float16* __restrict__ w2w,
    const _Float16* __restrict__ w3w,
    const float* __restrict__ b1, const float* __restrict__ b2,
    const float* __restrict__ gnAg, const float* __restrict__ gnAb,
    const float* __restrict__ gnVg,
    float* __restrict__ yaOut, float* __restrict__ yvOut, int M)
{
    __shared__ __align__(32) unsigned char smem[65536];
    _Float16* sXa  = (_Float16*)(smem);            // 2048h: input y_a tile (reused as sSA)
    _Float16* sSA  = (_Float16*)(smem);            // scratch A (aliases sXa)
    _Float16* sXv  = (_Float16*)(smem + 4096);     // 3*1024h: input y_v comps
    _Float16* sSB  = (_Float16*)(smem + 4096);     // scratch B (aliases sXv)
    _Float16* sL0  = (_Float16*)(smem + 10240);    // 2048h
    _Float16* sR0  = (_Float16*)(smem + 14336);    // 2048h
    _Float16* sL1  = (_Float16*)(smem + 18432);    // 3*2048h
    _Float16* sR1  = (_Float16*)(smem + 30720);    // 3*2048h
    float*    sYaT = (float*)(smem + 43008);       // 2048f: y_a_tp then y_a2
    float*    sYvT = (float*)(smem + 51200);       // 3*1024f
    float*    sStA = (float*)(smem + 63488);       // 16*8*2 f
    float*    sStV = (float*)(smem + 63488 + 2048 - 512); // 16*8 f (tail of spare)

    int tid  = threadIdx.x;
    int wave = tid >> 5, lane = tid & 31;
    int m0   = blockIdx.x * 16;
    int n0   = wave * 16;     // each wave owns columns n0..n0+15 (for N=128)
    int n    = lane & 15, hs = lane >> 4;

    // ---- stage inputs (f32 -> f16) -----------------------------------------
    for (int idx = tid; idx < 2048; idx += 256) {
        int m = idx >> 7, c = idx & 127;
        sXa[idx] = (_Float16)ya2[(long long)(m0 + m) * 128 + c];
    }
    for (int idx = tid; idx < 3072; idx += 256) {
        int i = idx >> 10, rem = idx & 1023, m = rem >> 6, c = rem & 63;
        sXv[idx] = (_Float16)yv2[((long long)(m0 + m) * 64 + c) * 3 + i];
    }
    __syncthreads();

    // ---- l0 = xa@L0^T, r0 = xa@R0^T ---------------------------------------
    {
        v8f aL = {}, aR = {};
        for (int k0 = 0; k0 < 128; k0 += 32) {
            v16h a = load_a_frag(sXa, 128, k0, lane);
            aL = wmma32(a, load_b_frag(L0w, 128, n0, k0, lane), aL);
            aR = wmma32(a, load_b_frag(R0w, 128, n0, k0, lane), aR);
        }
        store_tile_h(sL0, 128, aL, n0, lane);
        store_tile_h(sR0, 128, aR, n0, lane);
    }
    // ---- l1/r1 per component (K=64) ---------------------------------------
    for (int i = 0; i < 3; ++i) {
        v8f aL = {}, aR = {};
        for (int k0 = 0; k0 < 64; k0 += 32) {
            v16h a = load_a_frag(sXv + i * 1024, 64, k0, lane);
            aL = wmma32(a, load_b_frag(L1w, 64, n0, k0, lane), aL);
            aR = wmma32(a, load_b_frag(R1w, 64, n0, k0, lane), aR);
        }
        store_tile_h(sL1 + i * 2048, 128, aL, n0, lane);
        store_tile_h(sR1 + i * 2048, 128, aR, n0, lane);
    }
    __syncthreads();

    // ---- y_a = y000@Y000^T + y110@Y110^T ----------------------------------
    v8f accYa = {};
    for (int idx = tid; idx < 2048; idx += 256)        // y000 = l0*r0
        sSA[idx] = (_Float16)((float)sL0[idx] * (float)sR0[idx]);
    __syncthreads();
    for (int k0 = 0; k0 < 128; k0 += 32)
        accYa = wmma32(load_a_frag(sSA, 128, k0, lane),
                       load_b_frag(Y000w, 128, n0, k0, lane), accYa);
    __syncthreads();
    for (int idx = tid; idx < 2048; idx += 256) {      // y110 = sum_i l1*r1
        float s = 0.f;
#pragma unroll
        for (int i = 0; i < 3; ++i)
            s += (float)sL1[i * 2048 + idx] * (float)sR1[i * 2048 + idx];
        sSA[idx] = (_Float16)s;
    }
    __syncthreads();
    for (int k0 = 0; k0 < 128; k0 += 32)
        accYa = wmma32(load_a_frag(sSA, 128, k0, lane),
                       load_b_frag(Y110w, 128, n0, k0, lane), accYa);
    __syncthreads();
#pragma unroll
    for (int r = 0; r < 8; ++r) {                      // y_a_tp -> f32 + f16
        int m = r + 8 * hs;
        sYaT[m * 128 + n0 + n] = accYa[r];
        sSA [m * 128 + n0 + n] = (_Float16)accYa[r];
    }
    __syncthreads();

    // ---- MLP: h1 = leaky(ya@w1+b1); h2 = leaky(h1@w2+b2); ya2 += h2@w3 ----
    {
        v8f acc = {};
        for (int k0 = 0; k0 < 128; k0 += 32)
            acc = wmma32(load_a_frag(sSA, 128, k0, lane),
                         load_b_frag(w1w, 128, n0, k0, lane), acc);
#pragma unroll
        for (int r = 0; r < 8; ++r) {
            float v = acc[r] + b1[n0 + n];
            v = v > 0.f ? v : 0.1f * v;
            sSB[(r + 8 * hs) * 128 + n0 + n] = (_Float16)v;
        }
    }
    __syncthreads();
    {
        v8f acc = {};
        for (int k0 = 0; k0 < 128; k0 += 32)
            acc = wmma32(load_a_frag(sSB, 128, k0, lane),
                         load_b_frag(w2w, 128, n0, k0, lane), acc);
#pragma unroll
        for (int r = 0; r < 8; ++r) {
            float v = acc[r] + b2[n0 + n];
            v = v > 0.f ? v : 0.1f * v;
            sSA[(r + 8 * hs) * 128 + n0 + n] = (_Float16)v;
        }
    }
    __syncthreads();
    {
        v8f acc = {};
        for (int k0 = 0; k0 < 128; k0 += 32)
            acc = wmma32(load_a_frag(sSA, 128, k0, lane),
                         load_b_frag(w3w, 128, n0, k0, lane), acc);
#pragma unroll
        for (int r = 0; r < 8; ++r) {                  // y_a2 = y_a_tp + mlp
            int o = (r + 8 * hs) * 128 + n0 + n;
            float v = sYaT[o] + acc[r];
            sYaT[o] = v;
            sSB [o] = (_Float16)v;
        }
    }
    __syncthreads();

    // ---- l0 += ya2@updL^T ; r0 += ya2@updR^T ------------------------------
    {
        v8f aL = {}, aR = {};
        for (int k0 = 0; k0 < 128; k0 += 32) {
            v16h a = load_a_frag(sSB, 128, k0, lane);
            aL = wmma32(a, load_b_frag(updLw, 128, n0, k0, lane), aL);
            aR = wmma32(a, load_b_frag(updRw, 128, n0, k0, lane), aR);
        }
#pragma unroll
        for (int r = 0; r < 8; ++r) {
            int o = (r + 8 * hs) * 128 + n0 + n;
            sL0[o] = (_Float16)((float)sL0[o] + aL[r]);
            sR0[o] = (_Float16)((float)sR0[o] + aR[r]);
        }
    }
    __syncthreads();

    // ---- y_v[i] = y011@Y011^T + y101@Y101^T + y111@Y111^T  (N=64) ---------
    for (int i = 0; i < 3; ++i) {
        int i1 = (i + 1) % 3, i2 = (i + 2) % 3;
        v8f accV = {};
        for (int idx = tid; idx < 2048; idx += 256)    // y011 = l0 * r1[i]
            sSA[idx] = (_Float16)((float)sL0[idx] * (float)sR1[i * 2048 + idx]);
        __syncthreads();
        if (wave < 4)
            for (int k0 = 0; k0 < 128; k0 += 32)
                accV = wmma32(load_a_frag(sSA, 128, k0, lane),
                              load_b_frag(Y011w, 128, n0, k0, lane), accV);
        __syncthreads();
        for (int idx = tid; idx < 2048; idx += 256)    // y101 = l1[i] * r0
            sSA[idx] = (_Float16)((float)sL1[i * 2048 + idx] * (float)sR0[idx]);
        __syncthreads();
        if (wave < 4)
            for (int k0 = 0; k0 < 128; k0 += 32)
                accV = wmma32(load_a_frag(sSA, 128, k0, lane),
                              load_b_frag(Y101w, 128, n0, k0, lane), accV);
        __syncthreads();
        for (int idx = tid; idx < 2048; idx += 256)    // y111 = cross(l1,r1)[i]
            sSA[idx] = (_Float16)((float)sL1[i1 * 2048 + idx] * (float)sR1[i2 * 2048 + idx]
                                - (float)sL1[i2 * 2048 + idx] * (float)sR1[i1 * 2048 + idx]);
        __syncthreads();
        if (wave < 4) {
            for (int k0 = 0; k0 < 128; k0 += 32)
                accV = wmma32(load_a_frag(sSA, 128, k0, lane),
                              load_b_frag(Y111w, 128, n0, k0, lane), accV);
#pragma unroll
            for (int r = 0; r < 8; ++r)
                sYvT[i * 1024 + (r + 8 * hs) * 64 + n0 + n] = accV[r];
        }
        __syncthreads();
    }

    // ---- group norms ------------------------------------------------------
    if (tid < 128) {                                   // GN0 stats: (m, g) x 16ch
        int m = tid >> 3, g = tid & 7;
        float s = 0.f, ss = 0.f;
#pragma unroll
        for (int j = 0; j < 16; ++j) {
            float v = sYaT[m * 128 + g * 16 + j];
            s += v; ss += v * v;
        }
        float mean = s * (1.f / 16.f);
        float var  = ss * (1.f / 16.f) - mean * mean;
        sStA[(m * 8 + g) * 2 + 0] = mean;
        sStA[(m * 8 + g) * 2 + 1] = rsqrtf(var + 1e-5f);
    } else if (tid < 256) {                            // GN1 stats: (m, g) x 8ch x 3
        int t = tid - 128, m = t >> 3, g = t & 7;
        float ss = 0.f;
#pragma unroll
        for (int i = 0; i < 3; ++i)
#pragma unroll
            for (int j = 0; j < 8; ++j) {
                float v = sYvT[i * 1024 + m * 64 + g * 8 + j];
                ss += v * v;
            }
        sStV[m * 8 + g] = rsqrtf(ss * (1.f / 8.f) + 1e-5f);
    }
    __syncthreads();

    for (int idx = tid; idx < 2048; idx += 256) {      // y_a out = ya2 + GN0
        int m = idx >> 7, c = idx & 127, g = c >> 4;
        float mean = sStA[(m * 8 + g) * 2 + 0];
        float rs   = sStA[(m * 8 + g) * 2 + 1];
        float xn   = (sYaT[idx] - mean) * rs;
        long long row = m0 + m;
        yaOut[row * 128 + c] = ya2[row * 128 + c] + xn * gnAg[c] + gnAb[c];
    }
    for (int idx = tid; idx < 3072; idx += 256) {      // y_v out = yv2 + GN1
        int i = idx >> 10, rem = idx & 1023, m = rem >> 6, c = rem & 63;
        float rs = sStV[m * 8 + (c >> 3)];
        float v  = sYvT[i * 1024 + m * 64 + c] * rs * gnVg[c];
        long long row = m0 + m;
        long long o = (row * 64 + c) * 3 + i;
        yvOut[o] = yv2[o] + v;
    }
    (void)M;
}

// ---------------------------------------------------------------------------
// Small helper kernels
// ---------------------------------------------------------------------------
__global__ void cast_f16_kernel(const float* __restrict__ s, _Float16* __restrict__ d, int n) {
    int i = blockIdx.x * blockDim.x + threadIdx.x;
    if (i < n) d[i] = (_Float16)s[i];
}

// (O, Ci, Kw) -> [o][k*Ci + ci]
__global__ void pack_conv_kernel(const float* __restrict__ s, _Float16* __restrict__ d,
                                 int O, int Ci, int Kw) {
    int i = blockIdx.x * blockDim.x + threadIdx.x;
    int n = O * Ci * Kw;
    if (i >= n) return;
    int o = i / (Ci * Kw), rem = i - o * (Ci * Kw);
    int ci = rem / Kw, k = rem - ci * Kw;
    d[(long long)o * Ci * Kw + k * Ci + ci] = (_Float16)s[i];
}

// (A, D, O) -> [a][o][d]
__global__ void pack_ro_kernel(const float* __restrict__ s, _Float16* __restrict__ d,
                               int An, int Dn, int On) {
    int i = blockIdx.x * blockDim.x + threadIdx.x;
    int n = An * Dn * On;
    if (i >= n) return;
    int a = i / (Dn * On), rem = i - a * (Dn * On);
    int dd = rem / On, o = rem - dd * On;
    d[((long long)a * On + o) * Dn + dd] = (_Float16)s[i];
}

__global__ void centroid_kernel(const float* __restrict__ pos, float* __restrict__ pam, int n) {
    int idx = blockIdx.x * blockDim.x + threadIdx.x;
    if (idx >= n) return;                // n = B*NRES*3
    int i = idx % 3, r = idx / 3;        // r = b*NRES + n
    float s = 0.f;
#pragma unroll
    for (int a = 0; a < A_ATOMS; ++a) s += pos[((long long)r * A_ATOMS + a) * 3 + i];
    pam[idx] = s * (1.f / A_ATOMS);
}

// y_v += d0 + d1 + pe + nd   (in-place on yv_base, bndi layout)
__global__ void vec_prep_kernel(float* __restrict__ yv,
                                const float* __restrict__ pos0, const float* __restrict__ pos1,
                                const float* __restrict__ pam0, const float* __restrict__ pam1,
                                const float* __restrict__ enc0, const float* __restrict__ enc1,
                                const float* __restrict__ pew,
                                const float* __restrict__ ndl, const float* __restrict__ ndr,
                                int total) {
    int idx = blockIdx.x * blockDim.x + threadIdx.x;
    if (idx >= total) return;
    int i = idx % 3;
    int c = (idx / 3) & 63;
    int nn = (idx / 192) & 255;
    int b = idx / (192 * 256);
    long long r = (long long)b * NRES + nn;
    float p0 = pam0[r * 3 + i], p1 = pam1[r * 3 + i];
    float d0 = 0.f, d1 = 0.f;
#pragma unroll
    for (int a = 0; a < A_ATOMS; ++a) {
        long long at = (r * A_ATOMS + a) * 3 + i;
        d0 += enc0[c * A_ATOMS + a] * (pos0[at] - p0);
        d1 += enc1[c * A_ATOMS + a] * (pos1[at] - p1);
    }
    float pe = 0.1f * (p1 - p0) * pew[c];
    float nd = 0.f;
    if (nn < NRES - 1) {   // delta_l[n] = pos[n+1]-pos[n]
        float dA = pam0[(r + 1) * 3 + i] - p0;
        float dB = pam1[(r + 1) * 3 + i] - p1;
        nd += ndl[c * 2 + 0] * dA + ndl[c * 2 + 1] * dB;
    }
    if (nn > 0) {          // delta_r[n] = pos[n]-pos[n-1]
        float dA = p0 - pam0[(r - 1) * 3 + i];
        float dB = p1 - pam1[(r - 1) * 3 + i];
        nd += ndr[c * 2 + 0] * dA + ndr[c * 2 + 1] * dB;
    }
    yv[idx] += d0 + d1 + pe + nd;
}

// ---------------------------------------------------------------------------
// Host orchestration
// ---------------------------------------------------------------------------
extern "C" void kernel_launch(void* const* d_in, const int* in_sizes, int n_in,
                              void* d_out, int out_size, void* d_ws, size_t ws_size,
                              hipStream_t stream) {
    (void)in_sizes; (void)n_in; (void)out_size; (void)ws_size;
    // Inputs in setup_inputs() insertion order (params flattened in dict order):
    const float* x_a   = (const float*)d_in[0];
    const float* x_v   = (const float*)d_in[1];
    const float* y_a   = (const float*)d_in[2];
    const float* y_v   = (const float*)d_in[3];
    const float* pos0  = (const float*)d_in[4];
    const float* pos1  = (const float*)d_in[5];
    const float* ri_a  = (const float*)d_in[6];
    const float* ri_v  = (const float*)d_in[7];
    const float* enc0  = (const float*)d_in[8];
    const float* enc1  = (const float*)d_in[9];
    const float* pe_w  = (const float*)d_in[10];
    const float* nd_l  = (const float*)d_in[11];
    const float* nd_r  = (const float*)d_in[12];
    const float* ca1_w = (const float*)d_in[13];
    const float* ca1_b = (const float*)d_in[14];
    const float* ca2_w = (const float*)d_in[15];
    const float* ca2_b = (const float*)d_in[16];
    const float* ca3_w = (const float*)d_in[17];
    const float* ca3_b = (const float*)d_in[18];
    const float* cv_w  = (const float*)d_in[19];
    const float* L0p   = (const float*)d_in[20];
    const float* R0p   = (const float*)d_in[21];
    const float* L1p   = (const float*)d_in[22];
    const float* R1p   = (const float*)d_in[23];
    const float* updLp = (const float*)d_in[24];
    const float* updRp = (const float*)d_in[25];
    const float* Y000p = (const float*)d_in[26];
    const float* Y110p = (const float*)d_in[27];
    const float* Y011p = (const float*)d_in[28];
    const float* Y101p = (const float*)d_in[29];
    const float* Y111p = (const float*)d_in[30];
    const float* w1p   = (const float*)d_in[31];
    const float* b1p   = (const float*)d_in[32];
    const float* w2p   = (const float*)d_in[33];
    const float* b2p   = (const float*)d_in[34];
    const float* w3p   = (const float*)d_in[35];
    const float* gnAg  = (const float*)d_in[36];
    const float* gnAb  = (const float*)d_in[37];
    const float* gnVg  = (const float*)d_in[38];
    const float* ro_a  = (const float*)d_in[39];
    const float* ro_v  = (const float*)d_in[40];

    float* out = (float*)d_out;
    const long long oDxA = 0;
    const long long oDxV = (long long)B_DIM * NATOMS * DIM_A;               // 16777216
    const long long oYa  = oDxV + (long long)B_DIM * NATOMS * DIM_V * 3;    // 41943040
    const long long oYv  = oYa + (long long)MROWS * DIM_A;                  // 44040192
    float* yaFin = out + oYa;
    float* yvFin = out + oYv;

    // ---- workspace carving ------------------------------------------------
    char* wsb = (char*)d_ws;
    size_t off = 0;
    auto alloc = [&](size_t bytes) -> void* {
        void* p = wsb + off;
        off = (off + bytes + 255) & ~(size_t)255;
        return p;
    };
    float* yaBase = (float*)alloc((size_t)MROWS * DIM_A * 4);
    float* h1     = (float*)alloc((size_t)MROWS * DIM_A * 4);
    float* h2     = (float*)alloc((size_t)MROWS * DIM_A * 4);
    float* ya2    = (float*)alloc((size_t)MROWS * DIM_A * 4);
    float* yvBase = (float*)alloc((size_t)MROWS * DIM_V * 3 * 4);
    float* yv2    = (float*)alloc((size_t)MROWS * DIM_V * 3 * 4);
    float* pam0   = (float*)alloc((size_t)MROWS * 3 * 4);
    float* pam1   = (float*)alloc((size_t)MROWS * 3 * 4);
    _Float16* riAh  = (_Float16*)alloc(131072 * 2);
    _Float16* riVh  = (_Float16*)alloc(32768 * 2);
    _Float16* ca1h  = (_Float16*)alloc(81920 * 2);
    _Float16* ca2h  = (_Float16*)alloc(49152 * 2);
    _Float16* ca3h  = (_Float16*)alloc(49152 * 2);
    _Float16* cvh   = (_Float16*)alloc(36864 * 2);
    _Float16* L0h   = (_Float16*)alloc(16384 * 2);
    _Float16* R0h   = (_Float16*)alloc(16384 * 2);
    _Float16* L1h   = (_Float16*)alloc(8192 * 2);
    _Float16* R1h   = (_Float16*)alloc(8192 * 2);
    _Float16* updLh = (_Float16*)alloc(16384 * 2);
    _Float16* updRh = (_Float16*)alloc(16384 * 2);
    _Float16* Y000h = (_Float16*)alloc(16384 * 2);
    _Float16* Y110h = (_Float16*)alloc(16384 * 2);
    _Float16* Y011h = (_Float16*)alloc(8192 * 2);
    _Float16* Y101h = (_Float16*)alloc(8192 * 2);
    _Float16* Y111h = (_Float16*)alloc(8192 * 2);
    _Float16* w1h   = (_Float16*)alloc(16384 * 2);
    _Float16* w2h   = (_Float16*)alloc(16384 * 2);
    _Float16* w3h   = (_Float16*)alloc(16384 * 2);
    _Float16* roAh  = (_Float16*)alloc(131072 * 2);
    _Float16* roVh  = (_Float16*)alloc(32768 * 2);

    // ---- weight packing ---------------------------------------------------
    auto cast16 = [&](const float* s, _Float16* d, int n) {
        cast_f16_kernel<<<(n + 255) / 256, 256, 0, stream>>>(s, d, n);
    };
    cast16(ri_a, riAh, 131072);  cast16(ri_v, riVh, 32768);
    cast16(L0p, L0h, 16384);     cast16(R0p, R0h, 16384);
    cast16(L1p, L1h, 8192);      cast16(R1p, R1h, 8192);
    cast16(updLp, updLh, 16384); cast16(updRp, updRh, 16384);
    cast16(Y000p, Y000h, 16384); cast16(Y110p, Y110h, 16384);
    cast16(Y011p, Y011h, 8192);  cast16(Y101p, Y101h, 8192);  cast16(Y111p, Y111h, 8192);
    cast16(w1p, w1h, 16384);     cast16(w2p, w2h, 16384);     cast16(w3p, w3h, 16384);
    pack_conv_kernel<<<(81920 + 255) / 256, 256, 0, stream>>>(ca1_w, ca1h, 128, 128, 5);
    pack_conv_kernel<<<(49152 + 255) / 256, 256, 0, stream>>>(ca2_w, ca2h, 128, 128, 3);
    pack_conv_kernel<<<(49152 + 255) / 256, 256, 0, stream>>>(ca3_w, ca3h, 128, 128, 3);
    pack_conv_kernel<<<(36864 + 255) / 256, 256, 0, stream>>>(cv_w, cvh, 64, 64, 9);
    pack_ro_kernel<<<(131072 + 255) / 256, 256, 0, stream>>>(ro_a, roAh, 8, 128, 128);
    pack_ro_kernel<<<(32768 + 255) / 256, 256, 0, stream>>>(ro_v, roVh, 8, 64, 64);

    // ---- residue centroids ------------------------------------------------
    centroid_kernel<<<(MROWS * 3 + 255) / 256, 256, 0, stream>>>(pos0, pam0, MROWS * 3);
    centroid_kernel<<<(MROWS * 3 + 255) / 256, 256, 0, stream>>>(pos1, pam1, MROWS * 3);

    const int MT = MROWS / 16;  // 1024 row tiles

    // ---- readin: ya_base = y_a + 0.2 * (x_a @ ri_a^T)  [K=1024] -----------
    gemm_kernel<<<dim3(MT, 1, 1), 256, 0, stream>>>(
        x_a, 0LL, 1024, 1, 0LL, 0LL,
        riAh, 0LL,
        yaBase, 0LL, 128, 1, 0LL, 0LL,
        nullptr,
        y_a, 0LL, 128, 1, 0LL, 0LL,
        MROWS, 128, 1024, 0.2f, 0);

    // ---- readin: yv_base = y_v + 0.2 * (x_v[i] @ ri_v^T)  [K=512, z=i] ----
    gemm_kernel<<<dim3(MT, 1, 3), 256, 0, stream>>>(
        x_v, 0LL, 1536, 3, 0LL, 1LL,
        riVh, 0LL,
        yvBase, 0LL, 192, 3, 0LL, 1LL,
        nullptr,
        y_v, 0LL, 192, 3, 0LL, 1LL,
        MROWS, 64, 512, 0.2f, 0);

    // ---- yv_base += d0 + d1 + pe + nd -------------------------------------
    vec_prep_kernel<<<(MROWS * DIM_V * 3 + 255) / 256, 256, 0, stream>>>(
        yvBase, pos0, pos1, pam0, pam1, enc0, enc1, pe_w, nd_l, nd_r, MROWS * DIM_V * 3);

    // ---- chain convs ------------------------------------------------------
    conv_kernel<<<dim3(NRES / 16, B_DIM, 1), 256, 0, stream>>>(
        yaBase, 128, 1, (long long)NRES * 128, 0LL, ca1h,
        h1, 128, 1, (long long)NRES * 128, 0LL, ca1_b,
        nullptr, 0, 0, 0LL, 0LL,
        NRES, 128, 5, 128, 1);
    conv_kernel<<<dim3(NRES / 16, B_DIM, 1), 256, 0, stream>>>(
        h1, 128, 1, (long long)NRES * 128, 0LL, ca2h,
        h2, 128, 1, (long long)NRES * 128, 0LL, ca2_b,
        nullptr, 0, 0, 0LL, 0LL,
        NRES, 128, 3, 128, 1);
    conv_kernel<<<dim3(NRES / 16, B_DIM, 1), 256, 0, stream>>>(   // ya2 = yaBase + conv3
        h2, 128, 1, (long long)NRES * 128, 0LL, ca3h,
        ya2, 128, 1, (long long)NRES * 128, 0LL, ca3_b,
        yaBase, 128, 1, (long long)NRES * 128, 0LL,
        NRES, 128, 3, 128, 0);
    conv_kernel<<<dim3(NRES / 16, B_DIM, 3), 256, 0, stream>>>(   // yv2 = yvBase + cv
        yvBase, 192, 3, (long long)NRES * 192, 1LL, cvh,
        yv2, 192, 3, (long long)NRES * 192, 1LL, nullptr,
        yvBase, 192, 3, (long long)NRES * 192, 1LL,
        NRES, 64, 9, 64, 0);

    // ---- fused self tensor products + group norms + residuals -------------
    tp_kernel<<<MT, 256, 0, stream>>>(
        ya2, yv2,
        L0h, R0h, L1h, R1h, updLh, updRh,
        Y000h, Y110h, Y011h, Y101h, Y111h,
        w1h, w2h, w3h, b1p, b2p,
        gnAg, gnAb, gnVg,
        yaFin, yvFin, MROWS);

    // ---- readout: dx_a = 0.2 * y_a @ ro_a[a]^T  (grid.y = a) --------------
    gemm_kernel<<<dim3(MT, 8, 1), 256, 0, stream>>>(
        yaFin, 0LL, 128, 1, 0LL, 0LL,
        roAh, 16384LL,
        out + oDxA, 0LL, 1024, 1, 128LL, 0LL,
        nullptr,
        nullptr, 0LL, 0, 0, 0LL, 0LL,
        MROWS, 128, 128, 0.2f, 0);

    // ---- readout: dx_v = 0.2 * y_v[i] @ ro_v[a]^T  (grid.y=a, grid.z=i) ---
    gemm_kernel<<<dim3(MT, 8, 3), 256, 0, stream>>>(
        yvFin, 0LL, 192, 3, 0LL, 1LL,
        roVh, 4096LL,
        out + oDxV, 0LL, 1536, 3, 192LL, 1LL,
        nullptr,
        nullptr, 0LL, 0, 0, 0LL, 0LL,
        MROWS, 64, 64, 0.2f, 0);
}
